// TriangleAttention_80109730005202
// MI455X (gfx1250) — compile-verified, matrix operand order checked by reading, not run on hardware
//
#include <hip/hip_runtime.h>

// ---------------------------------------------------------------------------
// Triangle attention, fully fused, one workgroup per sequence-row n (B*L=320).
// bf16 WMMA (v_wmma_f32_16x16x32_bf16) for QKV / scores / PV / out-proj.
// CDNA5 data movement: global_load_async_to_lds_b128 (ASYNCcnt) for w_out
// staging, global_prefetch for residual panels, split-counter waits.
// LDS budget (dynamic, 313.3 KB):
//   Zn    [320][128] bf16  80 KB   normalized activations
//   Km    [320][128] bf16  80 KB   K (row-major: key x dh)
//   Vt    [128][320] bf16  80 KB   V transposed (dh x key)
//   Wo    [128][128] bf16  32 KB   staged w_out (async copy)
//   oScr  8 x [16][128]    32 KB   per-wave attention-output staging
//   pqScr 8 x [16][32]      8 KB   per-wave Q/P fragment re-layout scratch
//   biasL [320] f32       1.25 KB  key mask bias
// ---------------------------------------------------------------------------

typedef __attribute__((ext_vector_type(16))) __bf16 v16bf;
typedef __attribute__((ext_vector_type(8)))  float  v8f;

#define LL 320
#define DD 128
#define NHEAD 4
#define DH 32

// smem byte offsets (all 16B aligned)
#define OFF_ZN    0
#define OFF_KM    81920
#define OFF_VT    163840
#define OFF_WO    245760
#define OFF_OSCR  278528
#define OFF_PQSCR 311296
#define OFF_BIAS  319488
#define SMEM_BYTES 320768

static __device__ __forceinline__ unsigned short f2bf(float f) {
  union { float f; unsigned u; } x; x.f = f;
  unsigned u = x.u;
  unsigned r = u + 0x7FFFu + ((u >> 16) & 1u);   // round-to-nearest-even
  return (unsigned short)(r >> 16);
}

// Load one 16x32 bf16 A/B fragment (ISA 7.12.2 layout): for lane (m = lane&15,
// hi = lane>>4) the 16 elements are K = [8*hi .. 8*hi+7] and [16+8*hi .. +7],
// i.e. two contiguous 16-byte runs from row m of a row-major tile.
static __device__ __forceinline__ v16bf load_frag(const unsigned short* base,
                                                  int pitch, int lane) {
  int m  = lane & 15;
  int hi = (lane >> 4) & 1;
  const unsigned short* p = base + m * pitch + 8 * hi;
  union { v16bf v; uint4 u[2]; } f;
  f.u[0] = *(const uint4*)(p);
  f.u[1] = *(const uint4*)(p + 16);
  return f.v;
}

static __device__ __forceinline__ v8f wmma_bf16(v16bf a, v16bf b, v8f c) {
  return __builtin_amdgcn_wmma_f32_16x16x32_bf16(
      /*neg_a=*/false, a, /*neg_b=*/false, b,
      /*c_mod=*/(short)0, c, /*reuse_a=*/false, /*reuse_b=*/false);
}

static __device__ __forceinline__ float rsum16(float v) {
  v += __shfl_xor(v, 1); v += __shfl_xor(v, 2);
  v += __shfl_xor(v, 4); v += __shfl_xor(v, 8);
  return v;
}
static __device__ __forceinline__ float rmax16(float v) {
  v = fmaxf(v, __shfl_xor(v, 1)); v = fmaxf(v, __shfl_xor(v, 2));
  v = fmaxf(v, __shfl_xor(v, 4)); v = fmaxf(v, __shfl_xor(v, 8));
  return v;
}
static __device__ __forceinline__ float rsum32(float v) {
  v = rsum16(v); v += __shfl_xor(v, 16);
  return v;
}

// ---------------------------------------------------------------------------
// Prepass: convert w_qkv (384x128) and w_out (128x128) f32 -> bf16 into d_ws.
// Reused by all 320 workgroups; stays hot in the 192 MB L2.
// ---------------------------------------------------------------------------
__global__ void __launch_bounds__(256) cvt_weights_kernel(
    const float* __restrict__ wqkv, const float* __restrict__ wout,
    unsigned short* __restrict__ dst) {
  int i = blockIdx.x * 256 + threadIdx.x;          // 65536 threads total
  if (i < 3 * DD * DD) dst[i] = f2bf(wqkv[i]);
  int j = i - 3 * DD * DD;
  if (j >= 0 && j < DD * DD) dst[3 * DD * DD + j] = f2bf(wout[j]);
}

// ---------------------------------------------------------------------------
// Main fused kernel: grid = 320 (one block per n), block = 256 (8 wave32).
// ---------------------------------------------------------------------------
__global__ void __launch_bounds__(256) tri_attn_kernel(
    const float* __restrict__ z, const int* __restrict__ mask,
    const float* __restrict__ ln_g, const float* __restrict__ ln_b,
    const float* __restrict__ b_qkv, const float* __restrict__ b_out,
    const unsigned short* __restrict__ wqkv_bf,
    const unsigned short* __restrict__ wout_bf,
    float* __restrict__ out)
{
  extern __shared__ char smem[];
  unsigned short* Zn    = (unsigned short*)(smem + OFF_ZN);
  unsigned short* Km    = (unsigned short*)(smem + OFF_KM);
  unsigned short* Vt    = (unsigned short*)(smem + OFF_VT);
  unsigned short* Wo    = (unsigned short*)(smem + OFF_WO);
  unsigned short* oScr  = (unsigned short*)(smem + OFF_OSCR);
  unsigned short* pqScr = (unsigned short*)(smem + OFF_PQSCR);
  float*          biasL = (float*)(smem + OFF_BIAS);

  const int tid  = threadIdx.x;
  const int lane = tid & 31;
  const int wv   = tid >> 5;           // wave id 0..7
  const int ncol = lane & 15;
  const int hi   = (lane >> 4) & 1;
  const int nrow = blockIdx.x;         // sequence row n = 0..319
  const float* zb = z + (long)nrow * (LL * DD);

  // ---- stage w_out bf16 into LDS via async DMA (ASYNCcnt path) ----
  // Each lane moves one 16-byte packet per iteration; no VGPR data path.
  // LDS destination address = low 32 bits of the generic shared pointer.
  {
    unsigned lds_base = (unsigned)(unsigned long long)(const void*)Wo;
    for (int i = tid; i < (DD * DD) / 8; i += 256) {
      unsigned lds_addr = lds_base + (unsigned)(i * 16);
      unsigned long long ga =
          (unsigned long long)(const void*)wout_bf + (unsigned long long)i * 16;
      asm volatile("global_load_async_to_lds_b128 %0, %1, off"
                   :: "v"(lds_addr), "v"(ga) : "memory");
    }
  }

  // ---- key-mask bias (shared across all queries of this batch) ----
  for (int t = tid; t < LL; t += 256)
    biasL[t] = (mask[t] > 0) ? 0.0f : -1e30f;

  // ---- LayerNorm: one row per wave-iteration, 4 floats per lane ----
  {
    float4 g4 = *(const float4*)(ln_g + lane * 4);
    float4 b4 = *(const float4*)(ln_b + lane * 4);
    for (int row = wv; row < LL; row += 8) {
      float4 v = *(const float4*)(zb + row * DD + lane * 4);
      float s = v.x + v.y + v.z + v.w;
      float q = v.x * v.x + v.y * v.y + v.z * v.z + v.w * v.w;
      s = rsum32(s); q = rsum32(q);
      float mu   = s * (1.0f / DD);
      float var  = q * (1.0f / DD) - mu * mu;
      float rstd = rsqrtf(var + 1e-5f);
      float n0 = (v.x - mu) * rstd * g4.x + b4.x;
      float n1 = (v.y - mu) * rstd * g4.y + b4.y;
      float n2 = (v.z - mu) * rstd * g4.z + b4.z;
      float n3 = (v.w - mu) * rstd * g4.w + b4.w;
      uint2 pk;
      pk.x = (unsigned)f2bf(n0) | ((unsigned)f2bf(n1) << 16);
      pk.y = (unsigned)f2bf(n2) | ((unsigned)f2bf(n3) << 16);
      *(uint2*)(Zn + row * DD + lane * 4) = pk;   // ds_store_b64
    }
  }
  // async w_out copy must have landed before the barrier publishes Wo
  asm volatile("s_wait_asynccnt 0" ::: "memory");
  __syncthreads();

  // ---- K / V projection: 320 output 16x16 tiles, 40 per wave ----
  // tile id t: rt = t>>4 (row tile 0..19), ct = t&15 (0..7 -> K, 8..15 -> V)
  for (int ii = 0; ii < 40; ii++) {
    int t  = wv + 8 * ii;
    int rt = t >> 4;
    int ct = t & 15;
    int jrow = DD + ct * 16;            // qkv column base (128..383)
    v8f acc = {0, 0, 0, 0, 0, 0, 0, 0};
    #pragma unroll
    for (int kb = 0; kb < 4; kb++) {
      v16bf a = load_frag(Zn + rt * 16 * DD + kb * 32, DD, lane);
      v16bf b = load_frag(wqkv_bf + jrow * DD + kb * 32, DD, lane);
      acc = wmma_bf16(a, b, acc);
    }
    float bq = b_qkv[jrow + ncol];
    #pragma unroll
    for (int r = 0; r < 8; r++) acc[r] += bq;
    if (ct < 8) {                       // K: row-major [key][dh]
      int colk = ct * 16 + ncol;
      #pragma unroll
      for (int r = 0; r < 8; r++)
        Km[(rt * 16 + r + 8 * hi) * DD + colk] = f2bf(acc[r]);
    } else {                            // V: transposed [dh][key], pair-packed
      int dhc  = (ct - 8) * 16 + ncol;
      int keyb = rt * 16 + 8 * hi;
      #pragma unroll
      for (int r = 0; r < 8; r += 2) {
        unsigned pk = (unsigned)f2bf(acc[r]) | ((unsigned)f2bf(acc[r + 1]) << 16);
        *(unsigned*)(Vt + dhc * LL + keyb + r) = pk;   // ds_store_b32
      }
    }
  }
  __syncthreads();

  // ---- flash attention + out-projection; wave owns whole query tiles ----
  unsigned short* myO  = oScr  + wv * 16 * DD;
  unsigned short* myPQ = pqScr + wv * 16 * 32;
  const float qscale = 0.17677669529663687f;   // 1/sqrt(DH)

  for (int qt = wv; qt < 20; qt += 8) {
    // prefetch this tile's residual z panel (16 rows x 128 f32 = 8 KB):
    // 32 lanes x 256B; lowers to global_prefetch_b8, overlaps with attention
    __builtin_prefetch(zb + qt * 16 * DD + lane * 64, 0, 3);

    for (int h = 0; h < NHEAD; h++) {
      // Q tile (16x32) computed on the fly from Zn, re-laid out via scratch
      #pragma unroll
      for (int ct = 0; ct < 2; ct++) {
        v8f qa = {0, 0, 0, 0, 0, 0, 0, 0};
        #pragma unroll
        for (int kb = 0; kb < 4; kb++) {
          v16bf a = load_frag(Zn + qt * 16 * DD + kb * 32, DD, lane);
          v16bf b = load_frag(wqkv_bf + (h * DH + ct * 16) * DD + kb * 32, DD, lane);
          qa = wmma_bf16(a, b, qa);
        }
        float bq = b_qkv[h * DH + ct * 16 + ncol];
        #pragma unroll
        for (int r = 0; r < 8; r++)
          myPQ[(r + 8 * hi) * 32 + ct * 16 + ncol] = f2bf((qa[r] + bq) * qscale);
      }
      asm volatile("s_wait_dscnt 0" ::: "memory");
      v16bf qf = load_frag(myPQ, 32, lane);

      v8f o0 = {0, 0, 0, 0, 0, 0, 0, 0};
      v8f o1 = {0, 0, 0, 0, 0, 0, 0, 0};
      v8f mr = {-1e30f, -1e30f, -1e30f, -1e30f, -1e30f, -1e30f, -1e30f, -1e30f};
      v8f lr = {0, 0, 0, 0, 0, 0, 0, 0};

      for (int kc = 0; kc < 10; kc++) {           // 32 keys per chunk
        v16bf kb0 = load_frag(Km + (kc * 32)      * DD + h * DH, DD, lane);
        v16bf kb1 = load_frag(Km + (kc * 32 + 16) * DD + h * DH, DD, lane);
        v8f zz = {0, 0, 0, 0, 0, 0, 0, 0};
        v8f s0 = wmma_bf16(qf, kb0, zz);
        v8f s1 = wmma_bf16(qf, kb1, zz);
        float bk0 = biasL[kc * 32 + ncol];
        float bk1 = biasL[kc * 32 + 16 + ncol];
        #pragma unroll
        for (int r = 0; r < 8; r++) {
          float a0 = s0[r] + bk0;
          float a1 = s1[r] + bk1;
          float nm = fmaxf(mr[r], rmax16(fmaxf(a0, a1)));
          float fs = __expf(mr[r] - nm);
          mr[r] = nm;
          float p0 = __expf(a0 - nm);
          float p1 = __expf(a1 - nm);
          lr[r] = lr[r] * fs + rsum16(p0 + p1);
          o0[r] *= fs; o1[r] *= fs;
          myPQ[(r + 8 * hi) * 32 + ncol]      = f2bf(p0);
          myPQ[(r + 8 * hi) * 32 + 16 + ncol] = f2bf(p1);
        }
        asm volatile("s_wait_dscnt 0" ::: "memory");
        v16bf pf  = load_frag(myPQ, 32, lane);
        v16bf vb0 = load_frag(Vt + (h * DH)      * LL + kc * 32, LL, lane);
        v16bf vb1 = load_frag(Vt + (h * DH + 16) * LL + kc * 32, LL, lane);
        o0 = wmma_bf16(pf, vb0, o0);
        o1 = wmma_bf16(pf, vb1, o1);
      }
      // normalize and stage this head's 16x32 output slice
      #pragma unroll
      for (int r = 0; r < 8; r++) {
        float inv = 1.0f / lr[r];
        myO[(r + 8 * hi) * DD + h * DH + ncol]      = f2bf(o0[r] * inv);
        myO[(r + 8 * hi) * DD + h * DH + 16 + ncol] = f2bf(o1[r] * inv);
      }
    } // heads
    asm volatile("s_wait_dscnt 0" ::: "memory");

    // ---- out projection (16x128 @ 128x128) + bias + residual store ----
    for (int jt = 0; jt < 8; jt++) {
      v8f acc = {0, 0, 0, 0, 0, 0, 0, 0};
      #pragma unroll
      for (int kb = 0; kb < 4; kb++) {
        v16bf a = load_frag(myO + kb * 32, DD, lane);
        v16bf b = load_frag(Wo + (jt * 16) * DD + kb * 32, DD, lane);
        acc = wmma_bf16(a, b, acc);
      }
      int j = jt * 16 + ncol;
      float bo = b_out[j];
      #pragma unroll
      for (int r = 0; r < 8; r++) {
        int row = qt * 16 + r + 8 * hi;
        long off = (long)nrow * (LL * DD) + row * DD + j;
        out[off] = z[off] + acc[r] + bo;
      }
    }
  } // qt
}

extern "C" void kernel_launch(void* const* d_in, const int* in_sizes, int n_in,
                              void* d_out, int out_size, void* d_ws, size_t ws_size,
                              hipStream_t stream) {
  const float* z     = (const float*)d_in[0];
  const int*   mask  = (const int*)  d_in[1];
  const float* ln_g  = (const float*)d_in[2];
  const float* ln_b  = (const float*)d_in[3];
  const float* w_qkv = (const float*)d_in[4];
  const float* b_qkv = (const float*)d_in[5];
  const float* w_out = (const float*)d_in[6];
  const float* b_out = (const float*)d_in[7];
  float* out = (float*)d_out;

  unsigned short* wbf = (unsigned short*)d_ws;     // [3*128*128] qkv + [128*128] out

  cvt_weights_kernel<<<256, 256, 0, stream>>>(w_qkv, w_out, wbf);

  (void)hipFuncSetAttribute((const void*)tri_attn_kernel,
                            hipFuncAttributeMaxDynamicSharedMemorySize,
                            SMEM_BYTES);
  tri_attn_kernel<<<LL, 256, SMEM_BYTES, stream>>>(
      z, mask, ln_g, ln_b, b_qkv, b_out,
      wbf, wbf + 3 * DD * DD, out);
}
